// LlamaAttention_68865505624183
// MI455X (gfx1250) — matrix operand order repaired and checked by reading, not compiled, not run on hardware
//
#include <hip/hip_runtime.h>
#include <hip/hip_bf16.h>
#include <math.h>

typedef __attribute__((ext_vector_type(16))) __bf16 v16bf;
typedef __attribute__((ext_vector_type(8)))  __bf16 v8bf;
typedef __attribute__((ext_vector_type(8)))  float  v8f;

union V16U { v16bf v; v8bf h[2]; };

#define WMMA_BF16(a,b,c) \
  __builtin_amdgcn_wmma_f32_16x16x32_bf16(false,(a),false,(b),(short)0,(c),false,false)

// Problem constants (from reference)
constexpr int kB = 4, kS = 1024, kD = 4096;
constexpr int kNH = 32, kNKV = 8, kHD = 128;
constexpr int kBLOCK = 64, kNBLK = 16;
constexpr int kT = kB * kS;                      // 4096 tokens
constexpr int kTOTAL_BLOCKS = kB * kNBLK + 16;   // 80

// ---------------------------------------------------------------------------
// Tiled GEMM:  C[M,N] = A[M,K] (AT row-major -> bf16) * Bw[K,N] (f32 -> bf16)
// 128x128 WG tile, 8 waves (4x2 grid -> each wave 32x64), K in 32-chunks.
// LDS double-buffered; B staged TRANSPOSED so every WMMA fragment is a
// contiguous 16B/32B LDS vector load. v_wmma_f32_16x16x32_bf16 core.
// ---------------------------------------------------------------------------
template <typename AT, typename OT>
__global__ __launch_bounds__(256) void gemm_wmma(const AT* __restrict__ A,
                                                 const float* __restrict__ Bw,
                                                 OT* __restrict__ C,
                                                 int M, int N, int K) {
  // row stride 40 elems = 80B -> every 8-elem (16B) sub-row is 16B aligned
  __shared__ __attribute__((aligned(32))) __bf16 As[2][128][40];
  __shared__ __attribute__((aligned(32))) __bf16 Bst[2][128][40];  // [n][k]
  const int tid  = threadIdx.x;
  const int lane = tid & 31;
  const int wave = tid >> 5;       // 0..7
  const int wm   = wave >> 1;      // 0..3 (M dir)
  const int wn   = wave & 1;       // 0..1 (N dir)
  const int hl   = lane >> 4;
  const int mr   = lane & 15;
  const int m0   = blockIdx.y * 128;
  const int n0   = blockIdx.x * 128;
  // staging decomposition: A 128x32 (4 rows/thread), B 32x128 (4 rows/thread)
  const int ar = tid >> 3, ac = (tid & 7) * 4;
  const int br = tid >> 5, bc = (tid & 31) * 4;

  __bf16 aR[4][4], bR[4][4];
  // prologue: global load + LDS store of K-chunk 0
#pragma unroll
  for (int i = 0; i < 4; ++i)
#pragma unroll
    for (int j = 0; j < 4; ++j) {
      aR[i][j] = (__bf16)A[(size_t)(m0 + ar + 32 * i) * K + ac + j];
      bR[i][j] = (__bf16)Bw[(size_t)(br + 8 * i) * N + n0 + bc + j];
    }
#pragma unroll
  for (int i = 0; i < 4; ++i)
#pragma unroll
    for (int j = 0; j < 4; ++j) {
      As[0][ar + 32 * i][ac + j]  = aR[i][j];
      Bst[0][bc + j][br + 8 * i]  = bR[i][j];
    }
  __syncthreads();

  const int KT = K >> 5;
  v8f acc[2][4] = {};
  for (int kt = 0; kt < KT; ++kt) {
    const int cur = kt & 1;
    // issue next tile's global loads early (overlap with WMMA below)
    if (kt + 1 < KT) {
      const int k0 = (kt + 1) * 32;
#pragma unroll
      for (int i = 0; i < 4; ++i)
#pragma unroll
        for (int j = 0; j < 4; ++j) {
          aR[i][j] = (__bf16)A[(size_t)(m0 + ar + 32 * i) * K + k0 + ac + j];
          bR[i][j] = (__bf16)Bw[(size_t)(k0 + br + 8 * i) * N + n0 + bc + j];
        }
    }
    // fragments as contiguous LDS vector loads
    v16bf af[2], bfr[4];
#pragma unroll
    for (int mi = 0; mi < 2; ++mi) {
      int row = wm * 32 + mi * 16 + mr;
      V16U u;
      u.h[0] = *(const v8bf*)&As[cur][row][hl * 8];
      u.h[1] = *(const v8bf*)&As[cur][row][16 + hl * 8];
      af[mi] = u.v;
    }
#pragma unroll
    for (int ni = 0; ni < 4; ++ni) {
      int col = wn * 64 + ni * 16 + mr;
      V16U u;
      u.h[0] = *(const v8bf*)&Bst[cur][col][hl * 8];
      u.h[1] = *(const v8bf*)&Bst[cur][col][16 + hl * 8];
      bfr[ni] = u.v;
    }
#pragma unroll
    for (int mi = 0; mi < 2; ++mi)
#pragma unroll
      for (int ni = 0; ni < 4; ++ni)
        acc[mi][ni] = WMMA_BF16(af[mi], bfr[ni], acc[mi][ni]);
    // store next tile into the other LDS buffer
    if (kt + 1 < KT) {
      const int nxt = cur ^ 1;
#pragma unroll
      for (int i = 0; i < 4; ++i)
#pragma unroll
        for (int j = 0; j < 4; ++j) {
          As[nxt][ar + 32 * i][ac + j]  = aR[i][j];
          Bst[nxt][bc + j][br + 8 * i]  = bR[i][j];
        }
    }
    __syncthreads();
  }

  // C layout: M = hl*8 + r, N = mr
#pragma unroll
  for (int mi = 0; mi < 2; ++mi)
#pragma unroll
    for (int ni = 0; ni < 4; ++ni)
#pragma unroll
      for (int r = 0; r < 8; ++r) {
        int row = m0 + wm * 32 + mi * 16 + hl * 8 + r;
        int col = n0 + wn * 64 + ni * 16 + mr;
        C[(size_t)row * N + col] = (OT)acc[mi][ni][r];
      }
}

// ---------------------------------------------------------------------------
// RoPE on Q (bf16 in-place), folds in the 1/sqrt(HD) attention scale.
// ---------------------------------------------------------------------------
__global__ __launch_bounds__(256) void rope_q_kernel(__bf16* __restrict__ Q,
                                                     const int* __restrict__ pos_ids) {
  int idx = blockIdx.x * 256 + threadIdx.x;   // T*NH*64
  int d = idx & 63;
  int h = (idx >> 6) & 31;
  int t = idx >> 11;
  float p   = (float)pos_ids[t];
  float ang = p * __powf(10000.f, -(float)d * (1.f / 64.f));
  float cs = __cosf(ang), sn = __sinf(ang);
  const float sc = 0.08838834764831845f;      // 1/sqrt(128)
  size_t base = (size_t)t * (kNH * kHD) + h * kHD + d;
  float x1 = (float)Q[base], x2 = (float)Q[base + 64];
  Q[base]      = (__bf16)((x1 * cs - x2 * sn) * sc);
  Q[base + 64] = (__bf16)((x2 * cs + x1 * sn) * sc);
}

// RoPE K and scatter into paged bf16 cache [blk][off][hkv][hd]
__global__ __launch_bounds__(256) void rope_scatter_k_kernel(
    const __bf16* __restrict__ Kin, __bf16* __restrict__ kc,
    const int* __restrict__ pos_ids, const int* __restrict__ blk_off) {
  int idx = blockIdx.x * 256 + threadIdx.x;   // T*NKV*64
  int d = idx & 63;
  int h = (idx >> 6) & 7;
  int t = idx >> 9;
  int pos = pos_ids[t];
  int seq = t >> 10;
  int blk = blk_off[seq * kNBLK + (pos >> 6)];
  int off = pos & 63;
  float ang = (float)pos * __powf(10000.f, -(float)d * (1.f / 64.f));
  float cs = __cosf(ang), sn = __sinf(ang);
  size_t src = (size_t)t * (kNKV * kHD) + h * kHD + d;
  float x1 = (float)Kin[src], x2 = (float)Kin[src + 64];
  size_t dst = ((size_t)blk * kBLOCK + off) * (kNKV * kHD) + h * kHD + d;
  kc[dst]      = (__bf16)(x1 * cs - x2 * sn);
  kc[dst + 64] = (__bf16)(x2 * cs + x1 * sn);
}

// Scatter V into TRANSPOSED paged bf16 cache: vct[((blk*NKV+h)*HD+d)*64+off]
// (d-major -> attention's P*V B-fragments become contiguous 32B loads)
__global__ __launch_bounds__(256) void scatter_v_kernel(
    const __bf16* __restrict__ Vin, __bf16* __restrict__ vct,
    const int* __restrict__ pos_ids, const int* __restrict__ blk_off) {
  int idx = blockIdx.x * 256 + threadIdx.x;   // T*NKV*HD, off fastest
  int off = idx & 63;
  int d   = (idx >> 6) & 127;
  int h   = (idx >> 13) & 7;
  int sb  = idx >> 16;                        // b*16 + s_block
  int b   = sb >> 4;
  int s   = (sb & 15) * 64 + off;
  int t   = b * kS + s;
  int pos = pos_ids[t];
  int blk = blk_off[b * kNBLK + (pos >> 6)];
  vct[(((size_t)blk * kNKV + h) * kHD + d) * kBLOCK + (pos & 63)] =
      Vin[(size_t)t * (kNKV * kHD) + h * kHD + d];
}

// ---------------------------------------------------------------------------
// Flash attention (causal, GQA 4:1). Grid: B*NH*(S/64) WGs of 128 threads.
// Each wave owns 16 query rows; per 32-key step: preload 8 K fragments
// (contiguous 32B vector loads) -> 8 WMMA for Q*K^T, online softmax,
// preload V fragments (contiguous, transposed cache) -> 8 WMMA for P*V.
// Next tile prefetched via global_prefetch_b8.
// ---------------------------------------------------------------------------
__global__ __launch_bounds__(128) void attn_kernel(
    const __bf16* __restrict__ Qbf, const __bf16* __restrict__ kc,
    const __bf16* __restrict__ vct, const int* __restrict__ blk_off,
    __bf16* __restrict__ attn) {
  __shared__ __attribute__((aligned(32))) __bf16 plds[4][16][32];
  const int tid  = threadIdx.x;
  const int lane = tid & 31;
  const int wv   = tid >> 5;
  const int hl   = lane >> 4;
  const int mr   = lane & 15;
  const int bid  = blockIdx.x;               // b*NH*16 + h*16 + qt
  const int qt   = bid & 15;
  const int h    = (bid >> 4) & 31;
  const int b    = bid >> 9;
  const int hk   = h >> 2;                   // GQA: h / G
  const int s0   = qt * 64 + wv * 16;

  // Q fragments: two contiguous 16B loads each
  v16bf qf[4];
  const __bf16* qrow = Qbf + ((size_t)(b * kS + s0 + mr)) * (kNH * kHD) + h * kHD;
#pragma unroll
  for (int kk = 0; kk < 4; ++kk) {
    V16U u;
    u.h[0] = *(const v8bf*)(qrow + kk * 32 + hl * 16);
    u.h[1] = *(const v8bf*)(qrow + kk * 32 + hl * 16 + 8);
    qf[kk] = u.v;
  }

  v8f acc[8] = {};
  float mrow[8], lrow[8];
#pragma unroll
  for (int r = 0; r < 8; ++r) { mrow[r] = -3.0e38f; lrow[r] = 0.f; }

  const int s_hi = s0 + 15;
  for (int j = 0; j <= s_hi; j += 32) {
    int blk = blk_off[b * kNBLK + (j >> 6)];
    const __bf16* kbase = kc + ((size_t)blk * kBLOCK + (j & 63)) * (kNKV * kHD)
                             + hk * kHD + (size_t)mr * (kNKV * kHD);
    // preload all 8 K fragments (each one contiguous 32B per lane)
    v16bf kf[4][2];
#pragma unroll
    for (int kk = 0; kk < 4; ++kk) {
      kf[kk][0] = *(const v16bf*)(kbase + kk * 32 + hl * 16);
      kf[kk][1] = *(const v16bf*)(kbase + 16 * (kNKV * kHD) + kk * 32 + hl * 16);
    }
    // prefetch next 32-key tile (K rows + V rows) -> global_prefetch_b8
    if (j + 32 <= s_hi) {
      int blkn = blk_off[b * kNBLK + ((j + 32) >> 6)];
      const __bf16* kn = kc + ((size_t)blkn * kBLOCK + ((j + 32) & 63)) * (kNKV * kHD) + hk * kHD;
      __builtin_prefetch((const void*)(kn + (size_t)lane * (kNKV * kHD)), 0, 1);
      const __bf16* vn = vct + (((size_t)blkn * kNKV + hk) * kHD + lane * 4) * kBLOCK + ((j + 32) & 63);
      __builtin_prefetch((const void*)vn, 0, 1);
    }

    v8f c0 = {}, c1 = {};
#pragma unroll
    for (int kk = 0; kk < 4; ++kk) {
      c0 = WMMA_BF16(qf[kk], kf[kk][0], c0);
      c1 = WMMA_BF16(qf[kk], kf[kk][1], c1);
    }

    // online softmax over 32 columns; rows live in 16-lane halves
    const bool need_mask = (j + 31 > s0);
#pragma unroll
    for (int r = 0; r < 8; ++r) {
      int sq = s0 + hl * 8 + r;
      float v0 = c0[r], v1 = c1[r];
      if (need_mask) {
        if (j + mr > sq)      v0 = -3.0e38f;
        if (j + 16 + mr > sq) v1 = -3.0e38f;
      }
      float tmax = fmaxf(v0, v1);
#pragma unroll
      for (int o = 8; o > 0; o >>= 1) tmax = fmaxf(tmax, __shfl_xor(tmax, o, 32));
      float mnew = fmaxf(mrow[r], tmax);
      float p0 = __expf(v0 - mnew);
      float p1 = __expf(v1 - mnew);
      float rs = p0 + p1;
#pragma unroll
      for (int o = 8; o > 0; o >>= 1) rs += __shfl_xor(rs, o, 32);
      float alpha = __expf(mrow[r] - mnew);
      lrow[r] = lrow[r] * alpha + rs;
      mrow[r] = mnew;
#pragma unroll
      for (int ni = 0; ni < 8; ++ni) acc[ni][r] *= alpha;
      plds[wv][hl * 8 + r][mr]      = (__bf16)p0;
      plds[wv][hl * 8 + r][16 + mr] = (__bf16)p1;
    }

    // P as A-fragment (two contiguous 16B LDS loads)
    V16U up;
    up.h[0] = *(const v8bf*)&plds[wv][mr][hl * 16];
    up.h[1] = *(const v8bf*)&plds[wv][mr][hl * 16 + 8];
    v16bf pf = up.v;

    // O += P * V ; V fragments contiguous thanks to transposed cache
    const __bf16* vbb = vct + (((size_t)blk * kNKV + hk) * kHD) * kBLOCK + (j & 63) + hl * 16;
#pragma unroll
    for (int g = 0; g < 2; ++g) {
      v16bf vb[4];
#pragma unroll
      for (int q = 0; q < 4; ++q) {
        int ni = g * 4 + q;
        vb[q] = *(const v16bf*)(vbb + (size_t)(ni * 16 + mr) * kBLOCK);
      }
#pragma unroll
      for (int q = 0; q < 4; ++q)
        acc[g * 4 + q] = WMMA_BF16(pf, vb[q], acc[g * 4 + q]);
    }
  }

  // normalize + store bf16 [t, h*HD + d]
#pragma unroll
  for (int ni = 0; ni < 8; ++ni)
#pragma unroll
    for (int r = 0; r < 8; ++r) {
      float val = acc[ni][r] / lrow[r];
      size_t dst = ((size_t)(b * kS + s0 + hl * 8 + r)) * (kNH * kHD) + h * kHD + ni * 16 + mr;
      attn[dst] = (__bf16)val;
    }
}

// ---------------------------------------------------------------------------
extern "C" void kernel_launch(void* const* d_in, const int* in_sizes, int n_in,
                              void* d_out, int out_size, void* d_ws, size_t ws_size,
                              hipStream_t stream) {
  (void)in_sizes; (void)n_in; (void)out_size; (void)ws_size;
  const float* hidden = (const float*)d_in[0];
  const float* Wq     = (const float*)d_in[1];
  const float* Wk     = (const float*)d_in[2];
  const float* Wv     = (const float*)d_in[3];
  const float* Wo     = (const float*)d_in[4];
  const int* pos_ids  = (const int*)d_in[7];
  const int* blk_off  = (const int*)d_in[8];
  float* out = (float*)d_out;

  char* ws = (char*)d_ws;
  __bf16* Qbf  = (__bf16*)ws; ws += (size_t)kT * kNH * kHD * 2;                      // 32 MB
  __bf16* Kbf  = (__bf16*)ws; ws += (size_t)kT * kNKV * kHD * 2;                     //  8 MB
  __bf16* Vbf  = (__bf16*)ws; ws += (size_t)kT * kNKV * kHD * 2;                     //  8 MB
  __bf16* kcb  = (__bf16*)ws; ws += (size_t)kTOTAL_BLOCKS * kBLOCK * kNKV * kHD * 2; // 10.5 MB
  __bf16* vct  = (__bf16*)ws; ws += (size_t)kTOTAL_BLOCKS * kBLOCK * kNKV * kHD * 2; // 10.5 MB
  __bf16* attn = (__bf16*)ws; ws += (size_t)kT * kNH * kHD * 2;                      // 32 MB

  // QKV projections (bf16 WMMA, double-buffered)
  gemm_wmma<float, __bf16><<<dim3(32, 32), 256, 0, stream>>>(hidden, Wq, Qbf, kT, 4096, kD);
  gemm_wmma<float, __bf16><<<dim3(8, 32),  256, 0, stream>>>(hidden, Wk, Kbf, kT, 1024, kD);
  gemm_wmma<float, __bf16><<<dim3(8, 32),  256, 0, stream>>>(hidden, Wv, Vbf, kT, 1024, kD);

  // RoPE + paged cache scatter (V transposed)
  rope_q_kernel<<<(kT * kNH * 64) / 256, 256, 0, stream>>>(Qbf, pos_ids);
  rope_scatter_k_kernel<<<(kT * kNKV * 64) / 256, 256, 0, stream>>>(Kbf, kcb, pos_ids, blk_off);
  scatter_v_kernel<<<(kT * kNKV * 128) / 256, 256, 0, stream>>>(Vbf, vct, pos_ids, blk_off);

  // Causal GQA flash attention
  attn_kernel<<<kB * kNH * (kS / 64), 128, 0, stream>>>(Qbf, kcb, vct, blk_off, attn);

  // Output projection -> fp32 result
  gemm_wmma<__bf16, float><<<dim3(32, 32), 256, 0, stream>>>(attn, Wo, out, kT, 4096, kNH * kHD);
}
